// ContextAttention_29025388986793
// MI455X (gfx1250) — compile-verified, matrix operand order checked by reading, not run on hardware
//
#include <hip/hip_runtime.h>
#include <hip/hip_bf16.h>

// ---------------- problem constants ----------------
#define B_ 32
#define N_ 1025
#define D_ 768
#define H_ 12
#define d_ 64
#define K_ 257      // context length (R + T*U)
#define T_ 256
#define KPAD 288    // K padded to multiple of 32 for WMMA K-chunks
#define NT_Q 65     // ceil(1025/16) query tiles
#define MROWS ((long)B_ * N_)   // 32800 rows

typedef __attribute__((ext_vector_type(16))) _Float16 v16h;
typedef __attribute__((ext_vector_type(8)))  _Float16 v8h;
typedef __attribute__((ext_vector_type(8)))  float    v8f;

// D = A(16x32) x B(32x16) + C  via v_wmma_f32_16x16x32_f16
__device__ __forceinline__ v8f wmma32(v16h a, v16h b, v8f c) {
  return __builtin_amdgcn_wmma_f32_16x16x32_f16(
      /*neg_a=*/false, a, /*neg_b=*/false, b,
      /*c_mod=*/(short)0, c, /*reuse_a=*/false, /*reuse_b=*/false);
}

// Fragment loaders.  Lane (l = lane&15, hi = lane>>4) holds halves
// [k0 + 8*hi .. +7] and [k0 + 16 + 8*hi .. +7] of row `base` (ISA 7.12.2).
__device__ __forceinline__ v16h frag_f32(const float* base, bool valid, int k0, int hi) {
  v16h f;
  if (valid) {
    const float* p0 = base + k0 + hi * 8;
    const float* p1 = base + k0 + 16 + hi * 8;
#pragma unroll
    for (int i = 0; i < 8; ++i) { f[i] = (_Float16)p0[i]; f[i + 8] = (_Float16)p1[i]; }
  } else {
#pragma unroll
    for (int i = 0; i < 16; ++i) f[i] = (_Float16)0.f;
  }
  return f;
}

__device__ __forceinline__ v16h frag_f16v(const _Float16* base, bool valid, int k0, int hi) {
  v16h f;
  if (valid) {
    v8h u0 = *(const v8h*)(base + k0 + hi * 8);        // global_load_b128
    v8h u1 = *(const v8h*)(base + k0 + 16 + hi * 8);   // global_load_b128
#pragma unroll
    for (int i = 0; i < 8; ++i) { f[i] = u0[i]; f[i + 8] = u1[i]; }
  } else {
#pragma unroll
    for (int i = 0; i < 16; ++i) f[i] = (_Float16)0.f;
  }
  return f;
}

// 32-half LDS row, fragment slice for this lane
__device__ __forceinline__ v16h frag_lds_f16(const _Float16* rowp, int hi) {
  v16h f;
  v8h u0 = *(const v8h*)(rowp + hi * 8);               // ds_load_b128
  v8h u1 = *(const v8h*)(rowp + 16 + hi * 8);
#pragma unroll
  for (int i = 0; i < 8; ++i) { f[i] = u0[i]; f[i + 8] = u1[i]; }
  return f;
}

__device__ __forceinline__ v16h frag_lds_f32(const float* rowp, int k0, int hi) {
  v16h f;
#pragma unroll
  for (int i = 0; i < 8; ++i) {
    f[i]     = (_Float16)rowp[k0 + hi * 8 + i];
    f[i + 8] = (_Float16)rowp[k0 + 16 + hi * 8 + i];
  }
  return f;
}

// -------------------- 0. f32 -> f16 conversion (weights) --------------------
__global__ __launch_bounds__(256) void cvt_kernel(const float* __restrict__ src,
                                                  _Float16* __restrict__ dst, long n) {
  long i = (long)blockIdx.x * 256 + threadIdx.x;
  long stride = (long)gridDim.x * 256;
  for (; i < n; i += stride) dst[i] = (_Float16)src[i];
}

// -------------------- 1. tiled local pooling --------------------
// grid (B_, 257), block 128.  row 0: copy x[b,0,:]; row 1+t: softmax-pooled tile t.
__global__ __launch_bounds__(128) void pool_kernel(const float* __restrict__ x,
                                                   const float* __restrict__ Wlog,
                                                   _Float16* __restrict__ ctx) {
  int b = blockIdx.x;
  int row = blockIdx.y;
  int tid = threadIdx.x;
  const float* xb = x + (long)b * N_ * D_;
  _Float16* dst = ctx + ((long)b * K_ + row) * D_;
  if (row == 0) {
    for (int c = tid; c < D_; c += 128) dst[c] = (_Float16)xb[c];
    return;
  }
  __shared__ float part[4][128];
  __shared__ float wsm[4];
  int t = row - 1;
  int gy = t >> 4, gx = t & 15;
  long roff[4];
#pragma unroll
  for (int s = 0; s < 4; ++s) {
    int iy = s >> 1, ix = s & 1;
    roff[s] = (long)(1 + (gy * 2 + iy) * 32 + (gx * 2 + ix)) * D_;
  }
  float acc[4] = {0.f, 0.f, 0.f, 0.f};
  for (int c = tid; c < D_; c += 128) {
    float w = Wlog[c];
#pragma unroll
    for (int s = 0; s < 4; ++s) acc[s] += xb[roff[s] + c] * w;
  }
#pragma unroll
  for (int s = 0; s < 4; ++s) part[s][tid] = acc[s];
  __syncthreads();
  if (tid < 4) {
    float tot = 0.f;
    for (int i = 0; i < 128; ++i) tot += part[tid][i];
    part[tid][0] = tot;
  }
  __syncthreads();
  if (tid == 0) {
    float m = fmaxf(fmaxf(part[0][0], part[1][0]), fmaxf(part[2][0], part[3][0]));
    float e[4], sum = 0.f;
#pragma unroll
    for (int s = 0; s < 4; ++s) { e[s] = __expf(part[s][0] - m); sum += e[s]; }
#pragma unroll
    for (int s = 0; s < 4; ++s) wsm[s] = e[s] / sum;
  }
  __syncthreads();
  float w0 = wsm[0], w1 = wsm[1], w2 = wsm[2], w3 = wsm[3];
  for (int c = tid; c < D_; c += 128) {
    float v = w0 * xb[roff[0] + c] + w1 * xb[roff[1] + c] +
              w2 * xb[roff[2] + c] + w3 * xb[roff[3] + c];
    dst[c] = (_Float16)v;
  }
}

// -------------------- 2. CPB relative-position bias --------------------
__global__ __launch_bounds__(256) void cpb_kernel(const float* __restrict__ feats,
                                                  const float* __restrict__ maskp,
                                                  const float* __restrict__ w1,
                                                  const float* __restrict__ b1,
                                                  const float* __restrict__ w2,
                                                  const float* __restrict__ b2,
                                                  float* __restrict__ bias) {
  long idx = (long)blockIdx.x * 256 + threadIdx.x;
  if (idx >= (long)N_ * K_) return;
  int n = (int)(idx / K_);
  int kk = (int)(idx % K_);
  float f0 = feats[idx * 2 + 0];
  float f1 = feats[idx * 2 + 1];
  float msk = maskp[idx];
  float hbuf[32];
#pragma unroll
  for (int j = 0; j < 32; ++j) {
    float z = w1[j * 2 + 0] * f0 + w1[j * 2 + 1] * f1 + b1[j];
    hbuf[j] = 0.5f * z * (1.0f + erff(z * 0.70710678118654752f));
  }
#pragma unroll
  for (int hh = 0; hh < H_; ++hh) {
    float s = b2[hh];
#pragma unroll
    for (int j = 0; j < 32; ++j) s += w2[hh * 32 + j] * hbuf[j];
    bias[((long)hh * N_ + n) * K_ + kk] = s * msk;
  }
}

// -------------------- 3. Q projection: q = x @ Wq^T (f16 out) --------------------
// grid (2050, 12): wave computes 16 rows x 64 cols (A-frag reused across 4 B-frags)
__global__ __launch_bounds__(32) void gemm_q(const float* __restrict__ x,
                                             const _Float16* __restrict__ wqh,
                                             _Float16* __restrict__ qh) {
  int mt = blockIdx.x, nt = blockIdx.y;
  int lane = threadIdx.x, l = lane & 15, hi = lane >> 4;
  const float* arow = x + ((long)mt * 16 + l) * D_;
  const _Float16* w0 = wqh + ((long)nt * 64 + l) * D_;
  v8f acc0 = {}, acc1 = {}, acc2 = {}, acc3 = {};
  for (int k0 = 0; k0 < D_; k0 += 32) {
    if (k0 + 128 < D_) __builtin_prefetch(arow + k0 + 128, 0, 0);
    v16h a = frag_f32(arow, true, k0, hi);
    acc0 = wmma32(a, frag_f16v(w0 + 0L * 16 * D_, true, k0, hi), acc0);
    acc1 = wmma32(a, frag_f16v(w0 + 1L * 16 * D_, true, k0, hi), acc1);
    acc2 = wmma32(a, frag_f16v(w0 + 2L * 16 * D_, true, k0, hi), acc2);
    acc3 = wmma32(a, frag_f16v(w0 + 3L * 16 * D_, true, k0, hi), acc3);
  }
  long rbase = (long)mt * 16 + hi * 8;
  v8f accs[4] = {acc0, acc1, acc2, acc3};
#pragma unroll
  for (int c = 0; c < 4; ++c) {
    int col = nt * 64 + c * 16 + l;
#pragma unroll
    for (int i = 0; i < 8; ++i) qh[(rbase + i) * D_ + col] = (_Float16)accs[c][i];
  }
}

// -------------------- 4. KV projection with k / v^T scatter --------------------
// grid (18, 24, B_): wave 16 padded-ctx rows x 64 cols of [k | v].
__global__ __launch_bounds__(32) void gemm_kv(const _Float16* __restrict__ ctx,
                                              const _Float16* __restrict__ wkvh,
                                              _Float16* __restrict__ kbuf,
                                              _Float16* __restrict__ vT) {
  int mt = blockIdx.x, nt = blockIdx.y, b = blockIdx.z;
  int lane = threadIdx.x, l = lane & 15, hi = lane >> 4;
  int kk = mt * 16 + l;
  bool av = kk < K_;
  const _Float16* ap = ctx + ((long)b * K_ + (av ? kk : 0)) * D_;
  const _Float16* w0 = wkvh + ((long)nt * 64 + l) * D_;
  v8f acc0 = {}, acc1 = {}, acc2 = {}, acc3 = {};
  for (int k0 = 0; k0 < D_; k0 += 32) {
    v16h a = frag_f16v(ap, av, k0, hi);
    acc0 = wmma32(a, frag_f16v(w0 + 0L * 16 * D_, true, k0, hi), acc0);
    acc1 = wmma32(a, frag_f16v(w0 + 1L * 16 * D_, true, k0, hi), acc1);
    acc2 = wmma32(a, frag_f16v(w0 + 2L * 16 * D_, true, k0, hi), acc2);
    acc3 = wmma32(a, frag_f16v(w0 + 3L * 16 * D_, true, k0, hi), acc3);
  }
  int rbase = mt * 16 + hi * 8;
  v8f accs[4] = {acc0, acc1, acc2, acc3};
#pragma unroll
  for (int c = 0; c < 4; ++c) {
    int j = nt * 64 + c * 16 + l;      // 0..1535
#pragma unroll
    for (int i = 0; i < 8; ++i) {
      int row = rbase + i;             // padded ctx index 0..287
      _Float16 val = (_Float16)accs[c][i];
      if (j < D_) {                    // K half
        int hh = j >> 6, dd = j & 63;
        kbuf[((long)(b * H_ + hh) * KPAD + row) * d_ + dd] = val;
      } else {                         // V half, stored transposed [.., d, KPAD]
        int jj = j - D_;
        int hh = jj >> 6, dd = jj & 63;
        vT[((long)(b * H_ + hh) * d_ + dd) * KPAD + row] = val;
      }
    }
  }
}

// -------------------- 5. attention: QK^T * scale + bias, softmax, PV --------------------
// grid (65, H_, B_), one wave; logits/probs tile in LDS.
__global__ __launch_bounds__(32) void attn_kernel(const _Float16* __restrict__ qh,
                                                  const _Float16* __restrict__ kbuf,
                                                  const _Float16* __restrict__ vT,
                                                  const float* __restrict__ bias,
                                                  _Float16* __restrict__ oh) {
  __shared__ float sm[16][KPAD];
  int qt = blockIdx.x, h = blockIdx.y, b = blockIdx.z;
  int lane = threadIdx.x, l = lane & 15, hi = lane >> 4;
  int nbase = qt * 16;
  int qrow = nbase + l;
  bool qv = qrow < N_;
  const _Float16* qp = qh + ((long)b * N_ + (qv ? qrow : 0)) * D_ + h * d_;
  const _Float16* kb = kbuf + (long)(b * H_ + h) * KPAD * d_;

  // ---- phase 1: logits = q k^T ----
  v16h aq0 = frag_f16v(qp, qv, 0, hi);
  v16h aq1 = frag_f16v(qp, qv, 32, hi);
  for (int ct = 0; ct < 17; ++ct) {
    int kkb = ct * 16;
    bool kv = (kkb + l) < K_;
    const _Float16* kp = kb + (long)(kkb + (kv ? l : 0)) * d_;
    v8f acc = {};
    acc = wmma32(aq0, frag_f16v(kp, kv, 0, hi), acc);
    acc = wmma32(aq1, frag_f16v(kp, kv, 32, hi), acc);
#pragma unroll
    for (int i = 0; i < 8; ++i) sm[hi * 8 + i][kkb + l] = acc[i];
  }
  __syncthreads();

  // ---- phase 2: scale + bias + row softmax (lanes 0..15, one row each) ----
  if (lane < 16) {
    int row = lane;
    int n = nbase + row;
    if (n < N_) {
      const float* bp = bias + ((long)h * N_ + n) * K_;
      const float scale = 0.125f;   // 1/sqrt(64)
      float mx = -1e30f;
      for (int kk = 0; kk < K_; ++kk) {
        float v = sm[row][kk] * scale + bp[kk];
        sm[row][kk] = v;
        mx = fmaxf(mx, v);
      }
      float sum = 0.f;
      for (int kk = 0; kk < K_; ++kk) {
        float e = __expf(sm[row][kk] - mx);
        sm[row][kk] = e;
        sum += e;
      }
      float inv = 1.0f / sum;
      for (int kk = 0; kk < K_; ++kk) sm[row][kk] *= inv;
      for (int kk = K_; kk < KPAD; ++kk) sm[row][kk] = 0.f;
    } else {
      for (int kk = 0; kk < KPAD; ++kk) sm[row][kk] = 0.f;
    }
  }
  __syncthreads();

  // ---- phase 3: o = P @ V  (v stored transposed -> contiguous B rows) ----
  const _Float16* vb = vT + (long)(b * H_ + h) * d_ * KPAD;
  for (int dt = 0; dt < 4; ++dt) {
    const _Float16* vp = vb + (long)(dt * 16 + l) * KPAD;
    v8f acc = {};
    for (int kc = 0; kc < KPAD; kc += 32) {
      v16h a = frag_lds_f32(&sm[l][0], kc, hi);
      v16h bf = frag_f16v(vp, true, kc, hi);
      acc = wmma32(a, bf, acc);
    }
    int n0 = nbase + hi * 8;
#pragma unroll
    for (int i = 0; i < 8; ++i) {
      int n = n0 + i;
      if (n < N_) oh[((long)b * N_ + n) * D_ + h * d_ + dt * 16 + l] = (_Float16)acc[i];
    }
  }
}

// -------------------- 6. output projection with async-LDS W staging --------------------
// grid (513, 12), 4 waves.  Block tile 64 rows x 64 cols; the shared 64x32-half
// W chunk is staged into LDS with global_load_async_to_lds_b128 (ASYNCcnt).
__global__ __launch_bounds__(128) void gemm_o(const _Float16* __restrict__ oh,
                                              const _Float16* __restrict__ woh,
                                              const float* __restrict__ bo,
                                              float* __restrict__ out) {
  __shared__ _Float16 wtile[64 * 32];   // 4 KB, rows of 32 halves
  int mt = blockIdx.x, nt = blockIdx.y;
  int tid = threadIdx.x;
  int wave = tid >> 5;
  int lane = tid & 31, l = lane & 15, hi = lane >> 4;
  long row0 = (long)mt * 64 + wave * 16;
  bool mvalid = (row0 + l) < MROWS;
  const _Float16* ap = oh + (row0 + (mvalid ? l : 0)) * D_;
  // async staging assignment: 2 threads per W row, 16 halves (32B) each
  int srow = tid >> 1;
  int shalf = tid & 1;
  const _Float16* gsrc = woh + ((long)nt * 64 + srow) * D_ + shalf * 16;
  unsigned ldsoff = (unsigned)(uintptr_t)&wtile[srow * 32 + shalf * 16];

  v8f acc0 = {}, acc1 = {}, acc2 = {}, acc3 = {};
  for (int k0 = 0; k0 < D_; k0 += 32) {
    unsigned long long ga = (unsigned long long)(uintptr_t)(gsrc + k0);
    // copy 32 bytes global -> LDS via the async data mover path
    asm volatile("global_load_async_to_lds_b128 %0, %1, off"
                 :: "v"(ldsoff), "v"(ga) : "memory");
    asm volatile("global_load_async_to_lds_b128 %0, %1, off offset:16"
                 :: "v"(ldsoff), "v"(ga) : "memory");
    asm volatile("s_wait_asynccnt 0" ::: "memory");
    __syncthreads();

    v16h a = frag_f16v(ap, mvalid, k0, hi);
    acc0 = wmma32(a, frag_lds_f16(&wtile[(0 * 16 + l) * 32], hi), acc0);
    acc1 = wmma32(a, frag_lds_f16(&wtile[(1 * 16 + l) * 32], hi), acc1);
    acc2 = wmma32(a, frag_lds_f16(&wtile[(2 * 16 + l) * 32], hi), acc2);
    acc3 = wmma32(a, frag_lds_f16(&wtile[(3 * 16 + l) * 32], hi), acc3);
    __syncthreads();
  }
  v8f accs[4] = {acc0, acc1, acc2, acc3};
#pragma unroll
  for (int c = 0; c < 4; ++c) {
    int col = nt * 64 + c * 16 + l;
    float bv = bo[col];
    long rbase = row0 + hi * 8;
#pragma unroll
    for (int i = 0; i < 8; ++i) {
      long r = rbase + i;
      if (r < MROWS) out[r * D_ + col] = accs[c][i] + bv;
    }
  }
}

// -------------------- launcher --------------------
extern "C" void kernel_launch(void* const* d_in, const int* in_sizes, int n_in,
                              void* d_out, int out_size, void* d_ws, size_t ws_size,
                              hipStream_t stream) {
  const float* x      = (const float*)d_in[0];
  const float* Wlogit = (const float*)d_in[1];
  const float* Wq     = (const float*)d_in[2];
  const float* Wkv    = (const float*)d_in[3];
  const float* Wo     = (const float*)d_in[4];
  const float* bo     = (const float*)d_in[5];
  const float* cw1    = (const float*)d_in[6];
  const float* cb1    = (const float*)d_in[7];
  const float* cw2    = (const float*)d_in[8];
  const float* cb2    = (const float*)d_in[9];
  const float* feats  = (const float*)d_in[10];
  const float* maskp  = (const float*)d_in[11];
  float* out = (float*)d_out;

  char* ws = (char*)d_ws;
  size_t off = 0;
  auto take = [&](size_t bytes) -> char* {
    char* p = ws + off;
    off = (off + bytes + 255) & ~(size_t)255;
    return p;
  };
  _Float16* ctx  = (_Float16*)take((size_t)B_ * K_ * D_ * sizeof(_Float16));
  _Float16* qh   = (_Float16*)take((size_t)B_ * N_ * D_ * sizeof(_Float16));
  _Float16* kbuf = (_Float16*)take((size_t)B_ * H_ * KPAD * d_ * sizeof(_Float16));
  _Float16* vT   = (_Float16*)take((size_t)B_ * H_ * d_ * KPAD * sizeof(_Float16));
  float*    bias = (float*)   take((size_t)H_ * N_ * K_ * sizeof(float));
  _Float16* ohb  = (_Float16*)take((size_t)B_ * N_ * D_ * sizeof(_Float16));
  _Float16* wqh  = (_Float16*)take((size_t)D_ * D_ * sizeof(_Float16));
  _Float16* wkvh = (_Float16*)take((size_t)2 * D_ * D_ * sizeof(_Float16));
  _Float16* woh  = (_Float16*)take((size_t)D_ * D_ * sizeof(_Float16));
  (void)ws_size; (void)in_sizes; (void)n_in; (void)out_size;

  // 0. weight conversions (f32 -> f16, done once)
  cvt_kernel<<<1024, 256, 0, stream>>>(Wq, wqh, (long)D_ * D_);
  cvt_kernel<<<1024, 256, 0, stream>>>(Wkv, wkvh, (long)2 * D_ * D_);
  cvt_kernel<<<1024, 256, 0, stream>>>(Wo, woh, (long)D_ * D_);
  // 1. pooled context (f16)
  pool_kernel<<<dim3(B_, K_), 128, 0, stream>>>(x, Wlogit, ctx);
  // 2. CPB bias [H][N][K]
  {
    long total = (long)N_ * K_;
    int blocks = (int)((total + 255) / 256);
    cpb_kernel<<<blocks, 256, 0, stream>>>(feats, maskp, cw1, cb1, cw2, cb2, bias);
  }
  // 3. q = x @ Wq^T   (2050 row tiles x 12 col-quads)
  gemm_q<<<dim3(2050, 12), 32, 0, stream>>>(x, wqh, qh);
  // 4. kv = ctx @ Wkv^T, scatter k / v^T with zero padding rows
  gemm_kv<<<dim3(KPAD / 16, (2 * D_) / 64, B_), 32, 0, stream>>>(ctx, wkvh, kbuf, vT);
  // 5. attention
  attn_kernel<<<dim3(NT_Q, H_, B_), 32, 0, stream>>>(qh, kbuf, vT, bias, ohb);
  // 6. out = o @ Wo^T + bo (async-LDS staged W)
  gemm_o<<<dim3(513, 12), 128, 0, stream>>>(ohb, woh, bo, out);
}
